// DiscretizedMixture_68315749810344
// MI455X (gfx1250) — matrix operand order repaired and checked by reading, not compiled, not run on hardware
//
#include <hip/hip_runtime.h>
#include <hip/hip_bf16.h>
#include <math.h>

typedef _Float16 v16h __attribute__((ext_vector_type(16)));
typedef __fp16   p2h  __attribute__((ext_vector_type(2)));   // cvt_pkrtz result type
typedef float    v8f  __attribute__((ext_vector_type(8)));

#define KDIM   512
#define HWSZ   4096            // 64*64 pixels per image
#define NPAD   96              // 90 mixture params padded to 6 N-tiles of 16
#define NOUT   90
#define LDSB_BYTES   (16 * 6 * 512 * 2)          // 96 KB of f16 B fragments
#define LDSBIAS_OFF  (LDSB_BYTES)                // 96 floats
#define LDSC_OFF     (LDSB_BYTES + 96 * 4)       // 8 waves * 16 * 96 floats
#define SMEM_BYTES   (LDSC_OFF + 8 * 16 * 96 * 4)

__global__ __launch_bounds__(256)
void dmix_nll_kernel(const float* __restrict__ px_z,
                     const float* __restrict__ x,
                     const float* __restrict__ Wm,
                     const float* __restrict__ bias,
                     float* __restrict__ out)
{
    extern __shared__ char smem[];
    _Float16* ldsB   = (_Float16*)smem;                 // [kb][nt][32 lanes: N + 16*((K>>3)&1)][16 halves: by K]
    float*    ldsBias = (float*)(smem + LDSBIAS_OFF);
    float*    ldsC    = (float*)(smem + LDSC_OFF);      // [wave][m=16][n=96]

    const int tid  = threadIdx.x;
    const int wave = tid >> 5;
    const int lane = tid & 31;

    // ---- Stage W into LDS as f16 WMMA B fragments (coalesced global reads) ----
    for (int idx = tid; idx < NPAD * KDIM; idx += 256) {
        const int o = idx >> 9;          // output column 0..95
        const int k = idx & (KDIM - 1);
        const float wv = (o < NOUT) ? Wm[o * KDIM + k] : 0.0f;
        const int kb = k >> 5, kk = k & 31, nt = o >> 4, n = o & 15;
        // fragment layout: lane == 16*((K>>3)&1) + N, half s == (K&7) + 8*(K>>4)
        ldsB[(((kb * 6 + nt) * 32 + (((kk >> 3) & 1) << 4) + n) << 4) + (kk & 7) + ((kk >> 4) << 3)] = (_Float16)wv;
    }
    if (tid < 96) ldsBias[tid] = (tid < NOUT) ? bias[tid] : 0.0f;
    __syncthreads();

    // ---- GEMM: each wave owns one 16-pixel M tile, all 6 N tiles ----
    const long long P    = ((long long)blockIdx.x * 8 + wave) * 16;  // global pixel base
    const int       bimg = (int)(P >> 12);
    const int       hw   = (int)(P & (HWSZ - 1));
    // lanes 0-15: pixels hw..hw+15 (contiguous, coalesced), channel group +0
    // lanes 16-31: same pixels, channel group +8 (A-layout K pattern)
    const float* aptr = px_z + (size_t)bimg * KDIM * HWSZ + hw
                      + (lane & 15) + (size_t)((lane >> 4) * 8) * HWSZ;

    v8f acc[6] = {};
    for (int kb = 0; kb < 16; ++kb) {
        union { v16h v; p2h h[8]; } afrag;
#pragma unroll
        for (int v = 0; v < 8; ++v) {
            const int cb = (v < 4) ? (2 * v) : (2 * v + 8);  // K: 0..7 then 16..23
            const float f0 = aptr[(size_t)cb * HWSZ];
            const float f1 = aptr[(size_t)(cb + 1) * HWSZ];
            afrag.h[v] = __builtin_amdgcn_cvt_pkrtz(f0, f1); // v_cvt_pk f32->2xf16
        }
        aptr += (size_t)32 * HWSZ;
        if (kb < 15) __builtin_prefetch(aptr, 0, 0);         // global_prefetch_b8
#pragma unroll
        for (int nt = 0; nt < 6; ++nt) {
            const v16h bf = *(const v16h*)(ldsB + ((size_t)(kb * 6 + nt) << 9)
                                                + (lane << 4)); // 2x ds_load_b128
            acc[nt] = __builtin_amdgcn_wmma_f32_16x16x32_f16(
                false, afrag.v, false, bf, (short)0, acc[nt], false, false);
        }
    }

    // ---- Spill 16x96 f32 C tile to LDS (C layout: vgpr r -> M=r+8*(lane/16), N=lane%16) ----
    float* crow = ldsC + (size_t)wave * 16 * 96;
#pragma unroll
    for (int nt = 0; nt < 6; ++nt) {
#pragma unroll
        for (int r = 0; r < 8; ++r) {
            const int m = r + ((lane >> 4) << 3);
            const int n = nt * 16 + (lane & 15);
            crow[m * 96 + n] = acc[nt][r];
        }
    }
    __syncthreads();

    // ---- Mixture-NLL epilogue: 128 pixels * 3 channels per block ----
    const float inv_sqrt2 = 0.70710678118654752f;
    const float bin       = 0.5f / 127.5f;      // 1/STD/2
    for (int i = tid; i < 128 * 3; i += 256) {
        const int p = i / 3, c = i - p * 3;
        const long long Pg = (long long)blockIdx.x * 128 + p;
        const int bi2 = (int)(Pg >> 12);
        const int hw2 = (int)(Pg & (HWSZ - 1));
        const float xv = x[((size_t)bi2 * HWSZ + hw2) * 3 + c];
        const float* row = ldsC + (size_t)p * 96;

        float pl[10], maxl = -INFINITY;
#pragma unroll
        for (int m = 0; m < 10; ++m) {
            pl[m] = row[3 * m + c] + ldsBias[3 * m + c];
            maxl = fmaxf(maxl, pl[m]);
        }
        float sume = 0.0f, prob = 0.0f;
#pragma unroll
        for (int m = 0; m < 10; ++m) {
            const float mu = row[30 + 3 * m + c] + ldsBias[30 + 3 * m + c];
            const float ls = row[60 + 3 * m + c] + ldsBias[60 + 3 * m + c];
            const float sig = __expf(tanhf(ls * 0.125f) * 8.0f);
            const float e   = __expf(pl[m] - maxl);
            sume += e;
            const float inv = __frcp_rn(sig);
            const float zhi = (xv + bin - mu) * inv * inv_sqrt2;
            const float zlo = (xv - bin - mu) * inv * inv_sqrt2;
            prob += e * 0.5f * (erff(zhi) - erff(zlo));
        }
        out[(size_t)Pg * 3 + c] = -__logf(prob * __frcp_rn(sume) + 1e-8f);
    }
}

extern "C" void kernel_launch(void* const* d_in, const int* in_sizes, int n_in,
                              void* d_out, int out_size, void* d_ws, size_t ws_size,
                              hipStream_t stream) {
    const float* px_z = (const float*)d_in[0];   // [32, 512, 64, 64]
    const float* x    = (const float*)d_in[1];   // [32, 64, 64, 3]
    const float* Wm   = (const float*)d_in[2];   // [90, 512]
    const float* b    = (const float*)d_in[3];   // [90]
    float* out = (float*)d_out;                  // [32, 64, 64, 3]
    (void)in_sizes; (void)n_in; (void)out_size; (void)d_ws; (void)ws_size;

    // 131072 pixels / (8 waves * 16 pixels) = 1024 blocks
    dim3 grid(1024), block(256);
    dmix_nll_kernel<<<grid, block, SMEM_BYTES, stream>>>(px_z, x, Wm, b, out);
}
